// Model_Mil_67379446939986
// MI455X (gfx1250) — matrix operand order repaired and checked by reading, not compiled, not run on hardware
//
#include <hip/hip_runtime.h>
#include <hip/hip_bf16.h>
#include <math.h>

typedef _Float16 half_t;
typedef __attribute__((ext_vector_type(8)))  _Float16 v8h;
typedef __attribute__((ext_vector_type(16))) _Float16 v16h;
typedef __attribute__((ext_vector_type(8)))  float    v8f;

// Force global (addrspace 1) memory ops so the backend emits global_load_* /
// global_store_* instead of flat_*: FLAT ties up DScnt + the LDS path on CDNA5,
// and our inner loop needs LDS bandwidth for WMMA A-fragments.
#define GPTR(T, p) ((T __attribute__((address_space(1)))*)(p))

// ---------------- problem constants (from reference) ----------------
constexpr int BATCH = 1024;
constexpr int NSTEP = 100;
constexpr int DIMX  = 100;   // dim_x == dim_d
constexpr int DIMH  = 256;
constexpr int KIN   = 128;   // 101 zero-padded to 128
constexpr int NOUT  = 128;   // 100 zero-padded to 128

constexpr int MTILE    = 16;          // batch rows per workgroup
constexpr int NWAVE    = 8;           // wave32s per workgroup
constexpr int NTHREADS = NWAVE * 32;  // 256

constexpr int U_STRIDE = KIN  + 8;    // LDS half strides, padded vs bank conflicts
constexpr int H_STRIDE = DIMH + 8;
constexpr int X_STRIDE = 104;

// ---------------- workspace (d_ws) layout ----------------
constexpr int WIN_ELEMS  = DIMH * KIN;       // 32768 f16
constexpr int WHID_ELEMS = 3 * DIMH * DIMH;  // 196608 f16
constexpr int WOUT_ELEMS = NOUT * DIMH;      // 32768 f16
constexpr int WTOT = WIN_ELEMS + WHID_ELEMS + WOUT_ELEMS;
constexpr int BTOT = DIMH + 3 * DIMH + NOUT; // 1152 f32 biases

// ============ prep: convert weights/biases to padded f16/f32 in ws ============
__global__ void prep_weights(const float* __restrict__ Win,  const float* __restrict__ bin,
                             const float* __restrict__ Whid, const float* __restrict__ bhid,
                             const float* __restrict__ Wout, const float* __restrict__ bout,
                             half_t* __restrict__ wf, float* __restrict__ bf) {
  for (int i = blockIdx.x * blockDim.x + threadIdx.x; i < WTOT + BTOT;
       i += gridDim.x * blockDim.x) {
    if (i < WIN_ELEMS) {                       // W_in (256,101) -> (256,128), K zero-pad
      int n = i >> 7, k = i & 127;
      GPTR(half_t, wf)[i] = (half_t)((k < 101) ? GPTR(const float, Win)[n * 101 + k] : 0.0f);
    } else if (i < WIN_ELEMS + WHID_ELEMS) {   // W_hid (3,256,256) straight copy
      GPTR(half_t, wf)[i] = (half_t)GPTR(const float, Whid)[i - WIN_ELEMS];
    } else if (i < WTOT) {                     // W_out (100,256) -> (128,256), N zero-pad
      int j = i - WIN_ELEMS - WHID_ELEMS;
      int n = j >> 8, k = j & 255;
      GPTR(half_t, wf)[i] = (half_t)((n < 100) ? GPTR(const float, Wout)[n * 256 + k] : 0.0f);
    } else {                                   // biases
      int j = i - WTOT;
      float v;
      if (j < DIMH) v = GPTR(const float, bin)[j];
      else if (j < 4 * DIMH) v = GPTR(const float, bhid)[j - DIMH];
      else { int q = j - 4 * DIMH; v = (q < 100) ? GPTR(const float, bout)[q] : 0.0f; }
      GPTR(float, bf)[j] = v;
    }
  }
}

// ============ one 16xN GEMM layer on a wave (K multiples of 32) ============
// out[m,n] = act( sum_k A[m,k] * B[n,k] + bias[n] )
template <int KTILES, int NT, bool RELU, bool F16OUT>
__device__ __forceinline__ void gemm16(const half_t* __restrict__ A, int lda,
                                       const half_t* __restrict__ B, int ldb,
                                       const float* __restrict__ bias,
                                       half_t* Of16, float* Of32, int ldo,
                                       int lane, int wave) {
  // Launder the (loop-invariant) weight/bias pointers so the compiler cannot
  // hoist the weight loads out of the 100-step loop and spill them to scratch.
  // Weights are L2-resident (192MB L2, 0.5MB of weights shared by all WGs),
  // so re-streaming them from L2 each use is the cheap path on MI455X.
  asm volatile("" : "+s"(B), "+s"(bias));

  const v8f vzero = {0.f, 0.f, 0.f, 0.f, 0.f, 0.f, 0.f, 0.f};
  v8f acc[NT];
#pragma unroll
  for (int t = 0; t < NT; ++t) acc[t] = vzero;

  const int rowA    = lane & 15;
  const int halfSel = lane >> 4;  // 0: lanes 0-15, 1: lanes 16-31
  const int ncol    = lane & 15;
  const half_t* arow = A + rowA * lda;

#pragma unroll 2
  for (int kt = 0; kt < KTILES; ++kt) {
    const int k0 = kt * 32;
    // A fragment (CDNA5 16-bit A layout): lanes 0-15 hold K{k0..k0+7, k0+16..k0+23},
    // lanes 16-31 hold K{k0+8..k0+15, k0+24..k0+31}
    const int ka = k0 + halfSel * 8;
    v8h alo = *(const v8h*)(arow + ka);
    v8h ahi = *(const v8h*)(arow + ka + 16);
    v16h a = __builtin_shufflevector(alo, ahi, 0, 1, 2, 3, 4, 5, 6, 7,
                                     8, 9, 10, 11, 12, 13, 14, 15);
    // B fragment: lane = column n; same K layout as A (lanes 0-15 K{k0..k0+7, k0+16..k0+23}, lanes 16-31 K{k0+8..k0+15, k0+24..k0+31})
    const int kb = k0 + halfSel * 8;
#pragma unroll
    for (int t = 0; t < NT; ++t) {
      const int n = (wave * NT + t) * 16 + ncol;
      v8h blo = *GPTR(const v8h, B + (size_t)n * ldb + kb);
      v8h bhi = *GPTR(const v8h, B + (size_t)n * ldb + kb + 16);
      v16h b = __builtin_shufflevector(blo, bhi, 0, 1, 2, 3, 4, 5, 6, 7,
                                       8, 9, 10, 11, 12, 13, 14, 15);
      acc[t] = __builtin_amdgcn_wmma_f32_16x16x32_f16(
          false, a, false, b, (short)0, acc[t], false, false);
    }
  }
  // epilogue: C/D layout -> VGPR r, lane l: (m = r + 8*halfSel, n = l&15 + tile*16)
#pragma unroll
  for (int t = 0; t < NT; ++t) {
    const int n  = (wave * NT + t) * 16 + ncol;
    const float bv = *GPTR(const float, bias + n);
    const int rbase = halfSel * 8;
#pragma unroll
    for (int r = 0; r < 8; ++r) {
      float v = acc[t][r] + bv;
      if (RELU) v = fmaxf(v, 0.0f);
      const int m = rbase + r;
      if constexpr (F16OUT) Of16[m * ldo + n] = (half_t)v;
      else                  Of32[m * ldo + n] = v;
    }
  }
}

// ============ phi: 5-layer MLP, 16 batch rows, result f32 (16 x 128) ============
__device__ __forceinline__ void phi_eval(const half_t* u, half_t* h0, half_t* h1,
                                         float* out, const half_t* wf,
                                         const float* bf, int lane, int wave) {
  const half_t* Win  = wf;
  const half_t* Wh0  = wf + WIN_ELEMS;
  const half_t* Wh1  = Wh0 + DIMH * DIMH;
  const half_t* Wh2  = Wh1 + DIMH * DIMH;
  const half_t* Wout = wf + WIN_ELEMS + WHID_ELEMS;
  const float* bin = bf;
  const float* bh0 = bf + DIMH;
  const float* bh1 = bh0 + DIMH;
  const float* bh2 = bh1 + DIMH;
  const float* bou = bf + 4 * DIMH;

  gemm16<KIN / 32,  DIMH / (16 * NWAVE), true,  true >(u,  U_STRIDE, Win,  KIN,  bin, h0, nullptr, H_STRIDE, lane, wave);
  __syncthreads();
  gemm16<DIMH / 32, DIMH / (16 * NWAVE), true,  true >(h0, H_STRIDE, Wh0,  DIMH, bh0, h1, nullptr, H_STRIDE, lane, wave);
  __syncthreads();
  gemm16<DIMH / 32, DIMH / (16 * NWAVE), true,  true >(h1, H_STRIDE, Wh1,  DIMH, bh1, h0, nullptr, H_STRIDE, lane, wave);
  __syncthreads();
  gemm16<DIMH / 32, DIMH / (16 * NWAVE), true,  true >(h0, H_STRIDE, Wh2,  DIMH, bh2, h1, nullptr, H_STRIDE, lane, wave);
  __syncthreads();
  gemm16<DIMH / 32, NOUT / (16 * NWAVE), false, false>(h1, H_STRIDE, Wout, DIMH, bou, nullptr, out, NOUT, lane, wave);
  __syncthreads();
}

// ============ persistent per-batch-tile FBSDE kernel ============
__global__ __launch_bounds__(NTHREADS, 4) void fbsde_kernel(
    const float* __restrict__ W, const int* __restrict__ Sraw,
    const float* __restrict__ y0, const half_t* __restrict__ wf,
    const float* __restrict__ bf, float* __restrict__ out_x,
    float* __restrict__ out_y) {
  __shared__ __align__(16) half_t u_s[MTILE * U_STRIDE];
  __shared__ __align__(16) half_t h0_s[MTILE * H_STRIDE];
  __shared__ __align__(16) half_t h1_s[MTILE * H_STRIDE];
  __shared__ __align__(16) float  z_s[MTILE * NOUT];
  __shared__ __align__(16) float  e_s[MTILE * NOUT];
  __shared__ __align__(16) float  x_s[MTILE * X_STRIDE];
  __shared__ __align__(16) float  w_s[MTILE * X_STRIDE];
  __shared__ float y_s[MTILE];
  __shared__ float rbm[NTHREADS];
  __shared__ float rzw[NTHREADS];

  const int tid  = threadIdx.x;
  const int lane = tid & 31;
  const int wave = tid >> 5;
  const int m0   = blockIdx.x * MTILE;

  // init state: x = 100, y = y0; zero u padding (incl. cols 101..127, never touched again)
  for (int i = tid; i < MTILE * X_STRIDE; i += NTHREADS)
    x_s[i] = ((i % X_STRIDE) < DIMX) ? 100.0f : 0.0f;
  for (int i = tid; i < MTILE * U_STRIDE; i += NTHREADS) u_s[i] = (half_t)0.0f;
  if (tid < MTILE) y_s[tid] = GPTR(const float, y0)[0];
  __syncthreads();

  const float dt   = 1.0f / NSTEP;   // T = 1
  const float sdt  = sqrtf(dt);
  const float isdt = 1.0f / sdt;

  for (int n = 0; n < NSTEP; ++n) {
    const float t = dt * n;
    const float s = (float)(2 * GPTR(const int, Sraw)[n] - 1);

    // Brownian increments for this step: W[b][d][n]
    for (int i = tid; i < MTILE * DIMX; i += NTHREADS) {
      int m = i / DIMX, d = i % DIMX;
      w_s[m * X_STRIDE + d] =
          GPTR(const float, W)[((size_t)(m0 + m) * DIMX + d) * NSTEP + n];
    }
    // u = [x, t]
    for (int i = tid; i < MTILE * (DIMX + 1); i += NTHREADS) {
      int m = i / (DIMX + 1), d = i % (DIMX + 1);
      u_s[m * U_STRIDE + d] = (half_t)((d < DIMX) ? x_s[m * X_STRIDE + d] : t);
    }
    __syncthreads();
    phi_eval(u_s, h0_s, h1_s, z_s, wf, bf, lane, wave);   // z

    // u = [x + z*Xi, t]
    for (int i = tid; i < MTILE * DIMX; i += NTHREADS) {
      int m = i / DIMX, d = i % DIMX;
      float w    = w_s[m * X_STRIDE + d];
      float w2md = w * w - dt;
      float Xi   = 0.5f * isdt * w2md * s;
      u_s[m * U_STRIDE + d] =
          (half_t)(x_s[m * X_STRIDE + d] + z_s[m * NOUT + d] * Xi);
    }
    __syncthreads();
    phi_eval(u_s, h0_s, h1_s, e_s, wf, bf, lane, wave);   // netEval

    // per-row reductions: backwardMil = sum(netEval - z), zw = sum(z*w)
    {
      const int m = tid >> 4;   // 16 threads per row
      const int p = tid & 15;
      float bm = 0.0f, zw = 0.0f;
      for (int d = p; d < DIMX; d += 16) {
        float z = z_s[m * NOUT + d];
        bm += e_s[m * NOUT + d] - z;
        zw += z * w_s[m * X_STRIDE + d];
      }
      rbm[tid] = bm;
      rzw[tid] = zw;
    }
    __syncthreads();

    // x <- x + mu*x*dt + sig*x*w + 0.5*sig^2*(w^2 - dt)*x
    for (int i = tid; i < MTILE * DIMX; i += NTHREADS) {
      int m = i / DIMX, d = i % DIMX;
      float w   = w_s[m * X_STRIDE + d];
      float x   = x_s[m * X_STRIDE + d];
      float mil = 0.5f * 0.2f * 0.2f * (w * w - dt);
      x_s[m * X_STRIDE + d] = x + 0.05f * x * dt + 0.2f * x * w + mil * x;
    }
    // y <- y + r*y*dt + zw + backwardMil * s / sqrt(dt)
    if (tid < MTILE) {
      float bm = 0.0f, zw = 0.0f;
#pragma unroll
      for (int p = 0; p < 16; ++p) { bm += rbm[tid * 16 + p]; zw += rzw[tid * 16 + p]; }
      float y = y_s[tid];
      y_s[tid] = y + 0.05f * y * dt + zw + bm * s * isdt;
    }
    __syncthreads();
  }

  // write outputs: x flat (1024,100) then y (1024,1)
  for (int i = tid; i < MTILE * DIMX; i += NTHREADS) {
    int m = i / DIMX, d = i % DIMX;
    GPTR(float, out_x)[(size_t)(m0 + m) * DIMX + d] = x_s[m * X_STRIDE + d];
  }
  if (tid < MTILE) GPTR(float, out_y)[m0 + tid] = y_s[tid];
}

// ============ launch ============
extern "C" void kernel_launch(void* const* d_in, const int* in_sizes, int n_in,
                              void* d_out, int out_size, void* d_ws, size_t ws_size,
                              hipStream_t stream) {
  const float* W    = (const float*)d_in[0];   // (1024,100,100)
  const int*   Sraw = (const int*)d_in[1];     // (100,)
  const float* y0   = (const float*)d_in[2];   // (1,)
  const float* Win  = (const float*)d_in[3];   // (256,101)
  const float* bin  = (const float*)d_in[4];   // (256,)
  const float* Whid = (const float*)d_in[5];   // (3,256,256)
  const float* bhid = (const float*)d_in[6];   // (3,256)
  const float* Wout = (const float*)d_in[7];   // (100,256)
  const float* bout = (const float*)d_in[8];   // (100,)
  (void)in_sizes; (void)n_in; (void)out_size; (void)ws_size;

  half_t* wf = (half_t*)d_ws;                                   // 524288 B
  float*  bf = (float*)((char*)d_ws + (size_t)WTOT * sizeof(half_t));  // + 4608 B

  prep_weights<<<256, 256, 0, stream>>>(Win, bin, Whid, bhid, Wout, bout, wf, bf);

  float* out_x = (float*)d_out;
  float* out_y = out_x + (size_t)BATCH * DIMX;
  fbsde_kernel<<<BATCH / MTILE, NTHREADS, 0, stream>>>(W, Sraw, y0, wf, bf,
                                                       out_x, out_y);
}
